// GraphTransformer_9096740733074
// MI455X (gfx1250) — compile-verified, matrix operand order checked by reading, not run on hardware
//
#include <hip/hip_runtime.h>
#include <hip/hip_bf16.h>

// ---------------- problem constants ----------------
#define Bb   2048L
#define Nn   64L
#define Dd   512L
#define Hh   8L
#define HDhd 64L
#define Ll   3
#define FFff 2048L
#define BN   (Bb * Nn)   // 131072

typedef __attribute__((ext_vector_type(16))) _Float16 v16h;
typedef __attribute__((ext_vector_type(8)))  _Float16 v8h;
typedef __attribute__((ext_vector_type(8)))  float    v8f;

// =====================================================================
// Generic WMMA GEMM:  C[M,Nc] = act( A[M,K](f16,lda) @ W[K,Nc](f16,ldb) + bias )
// Block tile 128x128, K-step 32, 8 waves (wave32), each wave 32x64 (2x4 tiles).
// A tile staged with GLOBAL_LOAD_ASYNC_TO_LDS_B128 (ASYNCcnt path); B tile
// staged transposed via VALU (needs element reorder). act: 0=none,
// 1=leaky_relu(0.01), 2=relu. Writes f32 (Cf) and/or f16 (Ch).
// Requires M%128==0, Nc%128==0, K%32==0 (true for all uses here).
// =====================================================================
__global__ __launch_bounds__(256) void wmma_gemm(
    const _Float16* __restrict__ A, long lda,
    const _Float16* __restrict__ Bm, long ldb,
    const float* __restrict__ bias,
    float* __restrict__ Cf, _Float16* __restrict__ Ch, long ldc,
    int K, int act)
{
    __shared__ _Float16 Asub[128 * 32];   // [m][k]
    __shared__ _Float16 BsubT[128 * 32];  // [n][k]  (B transposed -> frag-ready)

    const int t    = threadIdx.x;
    const int lane = t & 31;
    const int wid  = t >> 5;
    const int wm   = (wid & 3) * 32;   // wave row offset inside block tile
    const int wn   = (wid >> 2) * 64;  // wave col offset inside block tile
    const int hlf  = lane >> 4;        // 0 or 1
    const int l16  = lane & 15;

    const long m0 = (long)blockIdx.y * 128;
    const long n0 = (long)blockIdx.x * 128;

    // per-thread staging coordinates (fixed across k-steps)
    const int arow  = t >> 1;
    const int akoff = (t & 1) * 16;
    const unsigned alds = (unsigned)(size_t)(void*)&Asub[arow * 32 + akoff];
    const int bn_  = t & 127;
    const int bkb  = (t >> 7) * 16;    // 0 or 16

    v8f acc[2][4] = {};

    for (int k0 = 0; k0 < K; k0 += 32) {
        // ---- stage A tile: 128x32, async cache->LDS, 2x16B per thread ----
        {
            const _Float16* gsrc = A + (m0 + arow) * lda + k0 + akoff;
            asm volatile("global_load_async_to_lds_b128 %0, %1, off"
                         :: "v"(alds), "v"(gsrc) : "memory");
            asm volatile("global_load_async_to_lds_b128 %0, %1, off offset:16"
                         :: "v"(alds), "v"(gsrc) : "memory");
        }
        // ---- stage B tile transposed: each thread gathers a 16-deep K column ----
        {
            v8h t0, t1;
#pragma unroll
            for (int i = 0; i < 8; ++i)
                t0[i] = Bm[(long)(k0 + bkb + i) * ldb + n0 + bn_];
#pragma unroll
            for (int i = 0; i < 8; ++i)
                t1[i] = Bm[(long)(k0 + bkb + 8 + i) * ldb + n0 + bn_];
            *(v8h*)(&BsubT[bn_ * 32 + bkb])     = t0;
            *(v8h*)(&BsubT[bn_ * 32 + bkb + 8]) = t1;
        }
        asm volatile("s_wait_asynccnt 0x0" ::: "memory");
        __syncthreads();

        // ---- build fragments (ISA 7.12.2 16-bit A 16x32 layout) ----
        v16h afrag[2], bfrag[4];
#pragma unroll
        for (int rt = 0; rt < 2; ++rt) {
            const _Float16* ap = &Asub[(wm + rt * 16 + l16) * 32 + hlf * 8];
            v8h lo = *(const v8h*)ap;
            v8h hi = *(const v8h*)(ap + 16);
            afrag[rt] = __builtin_shufflevector(lo, hi,
                0,1,2,3,4,5,6,7,8,9,10,11,12,13,14,15);
        }
#pragma unroll
        for (int ct = 0; ct < 4; ++ct) {
            const _Float16* bp = &BsubT[(wn + ct * 16 + l16) * 32 + hlf * 8];
            v8h lo = *(const v8h*)bp;
            v8h hi = *(const v8h*)(bp + 16);
            bfrag[ct] = __builtin_shufflevector(lo, hi,
                0,1,2,3,4,5,6,7,8,9,10,11,12,13,14,15);
        }
#pragma unroll
        for (int rt = 0; rt < 2; ++rt)
#pragma unroll
            for (int ct = 0; ct < 4; ++ct)
                acc[rt][ct] = __builtin_amdgcn_wmma_f32_16x16x32_f16(
                    false, afrag[rt], false, bfrag[ct],
                    (short)0, acc[rt][ct], false, false);
        __syncthreads();
    }

    // ---- epilogue: bias + activation + store ----
#pragma unroll
    for (int rt = 0; rt < 2; ++rt) {
#pragma unroll
        for (int ct = 0; ct < 4; ++ct) {
            const long col = n0 + wn + ct * 16 + l16;
            const float bvv = bias ? bias[col] : 0.0f;
#pragma unroll
            for (int i = 0; i < 8; ++i) {
                const long row = m0 + wm + rt * 16 + hlf * 8 + i;
                float v = acc[rt][ct][i] + bvv;
                if (act == 1) v = (v > 0.0f) ? v : v * 0.01f;
                else if (act == 2) v = (v > 0.0f) ? v : 0.0f;
                if (Cf) Cf[row * ldc + col] = v;
                if (Ch) Ch[row * ldc + col] = (_Float16)v;
            }
        }
    }
}

// =============== helper kernels ===============

__global__ void f32_to_f16_k(const float* __restrict__ s, _Float16* __restrict__ d, long n) {
    long i = (long)blockIdx.x * 256 + threadIdx.x;
    if (i < n) d[i] = (_Float16)s[i];
}

// (L,H,D,HD) f32 -> (L, D, H*HD) f16
__global__ void conv_wqkv_k(const float* __restrict__ s, _Float16* __restrict__ d) {
    long i = (long)blockIdx.x * 256 + threadIdx.x;
    if (i >= Ll * Hh * Dd * HDhd) return;
    long e  = i & 63;
    long dd = (i >> 6) & 511;
    long hh = (i >> 15) & 7;
    long l  = i >> 18;
    d[((l * Dd + dd) << 9) + (hh << 6) + e] = (_Float16)s[i];
}

// gather neighbor r/e rows into concat buffer segments [512..1023],[1024..1535]; neg mask
__global__ __launch_bounds__(256) void gather_nbr_k(
    const int* __restrict__ nbr_r, const int* __restrict__ nbr_e,
    const float* __restrict__ masks,
    const float* __restrict__ emb_e, const float* __restrict__ emb_r,
    _Float16* __restrict__ catb, float* __restrict__ neg)
{
    const long bn = blockIdx.x;
    const int  t  = threadIdx.x;
    const long rr = nbr_r[bn], ee = nbr_e[bn];
    _Float16* dst = catb + bn * 1536;
    const float* rs = emb_r + rr * Dd;
    const float* es = emb_e + ee * Dd;
    for (int d = t; d < 512; d += 256) {
        dst[512  + d] = (_Float16)rs[d];
        dst[1024 + d] = (_Float16)es[d];
    }
    if (t == 0) neg[bn] = 1e31f * (1.0f - masks[bn]);
}

// gather h = emb_e[e1], emb_q = emb_r[q]; also writes emb_q to output half 2
__global__ __launch_bounds__(256) void gather_hq_k(
    const int* __restrict__ e1, const int* __restrict__ qi,
    const float* __restrict__ emb_e, const float* __restrict__ emb_r,
    float* __restrict__ hf, _Float16* __restrict__ q16, float* __restrict__ outq)
{
    const long b = blockIdx.x;
    const int  t = threadIdx.x;
    const float* hs = emb_e + (long)e1[b] * Dd;
    const float* qs = emb_r + (long)qi[b] * Dd;
    for (int d = t; d < 512; d += 256) {
        hf[b * Dd + d] = hs[d];
        const float q = qs[d];
        q16[b * Dd + d]  = (_Float16)q;
        outq[b * Dd + d] = q;
    }
}

// broadcast current h (f32) into concat segment 0 as f16 for all N neighbors
__global__ void bcast_h_k(const float* __restrict__ hf, _Float16* __restrict__ catb) {
    long idx = (long)blockIdx.x * 256 + threadIdx.x;  // B*N*512 total
    long bn = idx >> 9;
    int  d  = (int)(idx & 511);
    long b  = bn >> 6;
    catb[bn * 1536 + d] = (_Float16)hf[b * Dd + d];
}

// per-batch attention: scores(H,N) -> softmax -> weighted V  (small dims, VALU)
__global__ __launch_bounds__(256) void attn_k(
    const float* __restrict__ qh, const _Float16* __restrict__ kh,
    const _Float16* __restrict__ vh, const float* __restrict__ neg,
    float* __restrict__ xout)
{
    const long b = blockIdx.x;
    const int  t = threadIdx.x;
    __shared__ float sc[512];
    __shared__ float at[512];
    const float inv_sqrt = 0.125f;  // 1/sqrt(64)

    for (int s = t; s < 512; s += 256) {
        const int hh = s >> 6, n = s & 63;
        const float*     q  = qh + b * 512 + hh * 64;
        const _Float16*  kp = kh + (b * 64 + n) * 512 + hh * 64;
        float d = 0.0f;
#pragma unroll 8
        for (int e = 0; e < 64; ++e) d += q[e] * (float)kp[e];
        sc[s] = d * inv_sqrt - neg[b * 64 + n];
    }
    __syncthreads();
    if (t < 8) {
        float mx = sc[t * 64];
        for (int n = 1; n < 64; ++n) mx = fmaxf(mx, sc[t * 64 + n]);
        float sum = 0.0f;
        for (int n = 0; n < 64; ++n) {
            const float e2 = expf(sc[t * 64 + n] - mx);
            at[t * 64 + n] = e2;
            sum += e2;
        }
        const float r = 1.0f / sum;
        for (int n = 0; n < 64; ++n) at[t * 64 + n] *= r;
    }
    __syncthreads();
    for (int c = t; c < 512; c += 256) {
        const int hh = c >> 6;
        float d = 0.0f;
#pragma unroll 8
        for (int n = 0; n < 64; ++n)
            d += at[hh * 64 + n] * (float)vh[(b * 64 + n) * 512 + c];
        xout[b * 512 + c] = d;
    }
}

// h = LayerNorm(h + x) * g + beta ; writes f32 and f16 copies of h
__global__ __launch_bounds__(256) void residual_ln_k(
    const float* __restrict__ x, const float* __restrict__ g, const float* __restrict__ bt,
    float* __restrict__ hf, _Float16* __restrict__ hh)
{
    const long b = blockIdx.x;
    const int  t = threadIdx.x;
    __shared__ float red[256];
    const float a0 = hf[b * 512 + t]       + x[b * 512 + t];
    const float a1 = hf[b * 512 + t + 256] + x[b * 512 + t + 256];
    red[t] = a0 + a1;
    __syncthreads();
    for (int s = 128; s > 0; s >>= 1) { if (t < s) red[t] += red[t + s]; __syncthreads(); }
    const float mean = red[0] * (1.0f / 512.0f);
    __syncthreads();
    const float d0 = a0 - mean, d1 = a1 - mean;
    red[t] = d0 * d0 + d1 * d1;
    __syncthreads();
    for (int s = 128; s > 0; s >>= 1) { if (t < s) red[t] += red[t + s]; __syncthreads(); }
    const float rstd = rsqrtf(red[0] * (1.0f / 512.0f) + 1e-5f);
    const float o0 = d0 * rstd * g[t]       + bt[t];
    const float o1 = d1 * rstd * g[t + 256] + bt[t + 256];
    hf[b * 512 + t]       = o0;
    hf[b * 512 + t + 256] = o1;
    hh[b * 512 + t]       = (_Float16)o0;
    hh[b * 512 + t + 256] = (_Float16)o1;
}

__global__ void copy_f32_k(const float* __restrict__ s, float* __restrict__ d, long n) {
    long i = (long)blockIdx.x * 256 + threadIdx.x;
    if (i < n) d[i] = s[i];
}

// =====================================================================
extern "C" void kernel_launch(void* const* d_in, const int* in_sizes, int n_in,
                              void* d_out, int out_size, void* d_ws, size_t ws_size,
                              hipStream_t stream)
{
    const int*   e1    = (const int*)d_in[0];
    const int*   qi    = (const int*)d_in[1];
    const int*   nbr_r = (const int*)d_in[2];
    const int*   nbr_e = (const int*)d_in[3];
    const float* masks = (const float*)d_in[4];
    const float* emb_e = (const float*)d_in[5];
    const float* emb_r = (const float*)d_in[6];
    const float* msg_W = (const float*)d_in[7];
    const float* msg_b = (const float*)d_in[8];
    const float* Wq    = (const float*)d_in[9];
    const float* bq    = (const float*)d_in[10];
    const float* Wk    = (const float*)d_in[11];
    const float* bk    = (const float*)d_in[12];
    const float* Wv    = (const float*)d_in[13];
    const float* bv    = (const float*)d_in[14];
    const float* fW1   = (const float*)d_in[15];
    const float* fb1   = (const float*)d_in[16];
    const float* fW2   = (const float*)d_in[17];
    const float* fb2   = (const float*)d_in[18];
    const float* ln1g  = (const float*)d_in[19];
    const float* ln1b  = (const float*)d_in[20];
    const float* ln2g  = (const float*)d_in[21];
    const float* ln2b  = (const float*)d_in[22];
    float* out = (float*)d_out;

    // ------- workspace carve-out -------
    char* p = (char*)d_ws;
    auto carve = [&](long bytes) -> void* {
        void* r = (void*)p;
        p += (bytes + 255) & ~255L;
        return r;
    };
    _Float16* catb  = (_Float16*)carve(BN * 1536 * 2);      // [h|r|e] concat, f16
    _Float16* value = (_Float16*)carve(BN * Dd * 2);
    _Float16* khb   = (_Float16*)carve(BN * Dd * 2);
    _Float16* vhb   = (_Float16*)carve(BN * Dd * 2);
    _Float16* msgW16= (_Float16*)carve(3 * Dd * Dd * 2);
    _Float16* WqT   = (_Float16*)carve(Ll * Dd * Dd * 2);
    _Float16* WkT   = (_Float16*)carve(Ll * Dd * Dd * 2);
    _Float16* WvT   = (_Float16*)carve(Ll * Dd * Dd * 2);
    _Float16* W1h   = (_Float16*)carve(Ll * Dd * FFff * 2);
    _Float16* W2h   = (_Float16*)carve(Ll * FFff * Dd * 2);
    _Float16* q16   = (_Float16*)carve(Bb * Dd * 2);
    _Float16* hh16  = (_Float16*)carve(Bb * Dd * 2);
    _Float16* ffmid = (_Float16*)carve(Bb * FFff * 2);
    float* hf    = (float*)carve(Bb * Dd * 4);
    float* qh    = (float*)carve(Bb * Dd * 4);
    float* neg   = (float*)carve(BN * 4);
    float* xbuf  = (float*)carve(Bb * Dd * 4);
    float* ffout = (float*)carve(Bb * Dd * 4);

    // ------- weight conversions (every call: deterministic, no caching) -------
    {
        long n = 3 * Dd * Dd;   // msg_W: (1536,512)
        f32_to_f16_k<<<dim3((n + 255) / 256), dim3(256), 0, stream>>>(msg_W, msgW16, n);
        long nq = Ll * Hh * Dd * HDhd;
        conv_wqkv_k<<<dim3((nq + 255) / 256), dim3(256), 0, stream>>>(Wq, WqT);
        conv_wqkv_k<<<dim3((nq + 255) / 256), dim3(256), 0, stream>>>(Wk, WkT);
        conv_wqkv_k<<<dim3((nq + 255) / 256), dim3(256), 0, stream>>>(Wv, WvT);
        long n1 = Ll * Dd * FFff;
        f32_to_f16_k<<<dim3((n1 + 255) / 256), dim3(256), 0, stream>>>(fW1, W1h, n1);
        f32_to_f16_k<<<dim3((n1 + 255) / 256), dim3(256), 0, stream>>>(fW2, W2h, n1);
    }

    // ------- gathers -------
    gather_hq_k<<<dim3((unsigned)Bb), dim3(256), 0, stream>>>(e1, qi, emb_e, emb_r,
                                                              hf, q16, out + Bb * Dd);
    gather_nbr_k<<<dim3((unsigned)BN), dim3(256), 0, stream>>>(nbr_r, nbr_e, masks,
                                                               emb_e, emb_r, catb, neg);
    bcast_h_k<<<dim3((unsigned)(BN * Dd / 256)), dim3(256), 0, stream>>>(hf, catb);

    const dim3 blk(256);
    const dim3 gBig(Dd / 128, BN / 128);     // (4,1024) : [131072 x 512]
    const dim3 gSm(Dd / 128, Bb / 128);      // (4,16)   : [2048 x 512]
    const dim3 gFF1(FFff / 128, Bb / 128);   // (16,16)  : [2048 x 2048]

    // value = leaky_relu(cat @ msg_W + msg_b)
    wmma_gemm<<<gBig, blk, 0, stream>>>(catb, 1536, msgW16, Dd, msg_b,
                                        nullptr, value, Dd, (int)(3 * Dd), 1);

    for (int l = 0; l < Ll; ++l) {
        // qh = emb_q @ Wq[l] + bq[l]      (f32 out)
        wmma_gemm<<<gSm, blk, 0, stream>>>(q16, Dd, WqT + (long)l * Dd * Dd, Dd,
                                           bq + (long)l * Hh * HDhd,
                                           qh, nullptr, Dd, (int)Dd, 0);
        // kh = r @ Wk[l] + bk[l]          (f16 out) ; A = r-segment of concat
        wmma_gemm<<<gBig, blk, 0, stream>>>(catb + 512, 1536, WkT + (long)l * Dd * Dd, Dd,
                                            bk + (long)l * Hh * HDhd,
                                            nullptr, khb, Dd, (int)Dd, 0);
        // vh = value @ Wv[l] + bv[l]      (f16 out)
        wmma_gemm<<<gBig, blk, 0, stream>>>(value, Dd, WvT + (long)l * Dd * Dd, Dd,
                                            bv + (long)l * Hh * HDhd,
                                            nullptr, vhb, Dd, (int)Dd, 0);
        // softmax attention
        attn_k<<<dim3((unsigned)Bb), blk, 0, stream>>>(qh, khb, vhb, neg, xbuf);
        // h = LN(h + x)
        residual_ln_k<<<dim3((unsigned)Bb), blk, 0, stream>>>(xbuf, ln1g + (long)l * Dd,
                                                              ln1b + (long)l * Dd, hf, hh16);
        // FFN
        wmma_gemm<<<gFF1, blk, 0, stream>>>(hh16, Dd, W1h + (long)l * Dd * FFff, FFff,
                                            fb1 + (long)l * FFff,
                                            nullptr, ffmid, FFff, (int)Dd, 2);
        wmma_gemm<<<gSm, blk, 0, stream>>>(ffmid, FFff, W2h + (long)l * FFff * Dd, Dd,
                                           fb2 + (long)l * Dd,
                                           ffout, nullptr, Dd, (int)FFff, 0);
        residual_ln_k<<<dim3((unsigned)Bb), blk, 0, stream>>>(ffout, ln2g + (long)l * Dd,
                                                              ln2b + (long)l * Dd, hf, hh16);
        // recompute value = msg_enc(h) for next layer (skip after last: unused)
        if (l < Ll - 1) {
            bcast_h_k<<<dim3((unsigned)(BN * Dd / 256)), dim3(256), 0, stream>>>(hf, catb);
            wmma_gemm<<<gBig, blk, 0, stream>>>(catb, 1536, msgW16, Dd, msg_b,
                                                nullptr, value, Dd, (int)(3 * Dd), 1);
        }
    }

    // out[0 : B*D] = h
    copy_f32_k<<<dim3((unsigned)(Bb * Dd / 256)), blk, 0, stream>>>(hf, out, Bb * Dd);
}
